// DecoderGRU_85959475462669
// MI455X (gfx1250) — compile-verified, matrix operand order checked by reading, not run on hardware
//
#include <hip/hip_runtime.h>
#include <hip/hip_bf16.h>
#include <math.h>

typedef __attribute__((ext_vector_type(16))) _Float16 v16h;
typedef __attribute__((ext_vector_type(8)))  float    v8f;

// ---------------------------------------------------------------------------
// Fragment loader for v_wmma_f32_16x16x32_f16.
// 16-bit A (16x32) layout: ln = lane&15 -> row M (col N for B), hi = lane>>4;
//   elements 0..7  <-> K = k0 + 8*hi + j
//   elements 8..15 <-> K = k0 + 16 + 8*hi + j
// Both chunks are 8 contiguous f32 in memory -> b128-pair loads, then
// in-register v_cvt_pk_f16_f32 (HBM traffic stays at the f32 source size).
// ---------------------------------------------------------------------------
__device__ __forceinline__ v16h load_frag_f32(const float* __restrict__ p0,
                                              const float* __restrict__ p1) {
  v16h f;
#pragma unroll
  for (int j = 0; j < 8; ++j) {
    f[j]     = (_Float16)p0[j];
    f[8 + j] = (_Float16)p1[j];
  }
  return f;
}

// ---------------------------------------------------------------------------
// Generic C[M,N] = A[M,K] * W[N,K]^T + bias[N].
// One wave per 16x64 output tile: 4 independent WMMA accumulator chains
// (hides the WMMA->WMMA RAW hazard), A-fragment amortized over 4 WMMAs.
// grid.x = N/64, grid.y = M/16, block = 32 (one wave, EXEC all ones).
// ---------------------------------------------------------------------------
__global__ __launch_bounds__(32) void gemm_wmma_f16x4(
    const float* __restrict__ A, int lda,
    const float* __restrict__ W, int ldw,
    const float* __restrict__ bias,
    float* __restrict__ C, int ldc, int K) {
  const int lane = threadIdx.x;
  const int ln = lane & 15;
  const int hi = lane >> 4;
  const int n0 = blockIdx.x * 64;
  const int m0 = blockIdx.y * 16;

  const float* __restrict__ arow = A + (long)(m0 + ln) * lda;
  const float* wrow[4];
#pragma unroll
  for (int t = 0; t < 4; ++t)
    wrow[t] = W + (long)(n0 + t * 16 + ln) * ldw;

  v8f c[4];
#pragma unroll
  for (int t = 0; t < 4; ++t) c[t] = (v8f){};

  for (int k0 = 0; k0 < K; k0 += 32) {
    // Prefetch the streamed weight rows ahead (emits global_prefetch_b8;
    // no counter tracking, pure latency hiding for the 23.3 TB/s stream).
    if (k0 + 128 < K) {
#pragma unroll
      for (int t = 0; t < 4; ++t)
        __builtin_prefetch(wrow[t] + k0 + 128, 0, 1);
    }
    v16h a = load_frag_f32(arow + k0 + 8 * hi, arow + k0 + 16 + 8 * hi);
#pragma unroll
    for (int t = 0; t < 4; ++t) {
      v16h b = load_frag_f32(wrow[t] + k0 + 8 * hi,
                             wrow[t] + k0 + 16 + 8 * hi);
      c[t] = __builtin_amdgcn_wmma_f32_16x16x32_f16(false, a, false, b,
                                                    (short)0, c[t],
                                                    false, false);
    }
  }
  // C/D layout: element i <-> M = m0 + i + 8*hi, N = n + ln (coalesced over N)
#pragma unroll
  for (int t = 0; t < 4; ++t) {
    const int n = n0 + t * 16 + ln;
    const float bn = bias[n];
#pragma unroll
    for (int i = 0; i < 8; ++i)
      C[(long)(m0 + i + 8 * hi) * ldc + n] = c[t][i] + bn;
  }
}

// ---------------------------------------------------------------------------
// Fused attention scores:
//   scores[m] = sum_n V[n] * tanh( (enc @ Ww^T)[m,n] + Wb[n] + Uh[b,n] ) + Vb
// One wave per 16-row tile of enc (B*S/16 = 1024 waves). N=512 processed as
// 4 groups of 8 WMMA tiles held in registers; energy tensor never hits memory.
// ---------------------------------------------------------------------------
__global__ __launch_bounds__(32) void attn_scores_wmma(
    const float* __restrict__ enc,   // [B*S, 512] row-major
    const float* __restrict__ Ww,    // [512, 512] row-major
    const float* __restrict__ Wb,    // [512]
    const float* __restrict__ Uh,    // [B, 512]
    const float* __restrict__ Vw,    // [512]
    const float* __restrict__ Vb,    // [1]
    float* __restrict__ scores) {    // [B*S]
  const int lane = threadIdx.x;
  const int ln = lane & 15;
  const int hi = lane >> 4;
  const int m0 = blockIdx.x * 16;          // global row tile (b*S + s)
  const int b  = m0 >> 8;                  // S = 256, 256 % 16 == 0
  const float* __restrict__ arow = enc + (long)(m0 + ln) * 512;
  const float* __restrict__ uh   = Uh + (long)b * 512;

  float srow[8];
#pragma unroll
  for (int i = 0; i < 8; ++i) srow[i] = 0.0f;

  for (int g = 0; g < 4; ++g) {            // 4 groups x 128 cols
    v8f c[8];
#pragma unroll
    for (int t = 0; t < 8; ++t) c[t] = (v8f){};

    for (int k0 = 0; k0 < 512; k0 += 32) {
      v16h a = load_frag_f32(arow + k0 + 8 * hi, arow + k0 + 16 + 8 * hi);
#pragma unroll
      for (int t = 0; t < 8; ++t) {
        const int n = g * 128 + t * 16 + ln;
        const float* __restrict__ wrow = Ww + (long)n * 512;
        v16h bf = load_frag_f32(wrow + k0 + 8 * hi, wrow + k0 + 16 + 8 * hi);
        c[t] = __builtin_amdgcn_wmma_f32_16x16x32_f16(false, a, false, bf,
                                                      (short)0, c[t],
                                                      false, false);
      }
    }
    // tanh + V-dot, accumulate per-lane partial score for rows m0 + i + 8*hi
#pragma unroll
    for (int t = 0; t < 8; ++t) {
      const int n = g * 128 + t * 16 + ln;
      const float add = Wb[n] + uh[n];
      const float v = Vw[n];
#pragma unroll
      for (int i = 0; i < 8; ++i)
        srow[i] += v * tanhf(c[t][i] + add);
    }
  }
  // Reduce over the 16 lanes of each half-wave (xor <= 8 stays in-half).
#pragma unroll
  for (int off = 8; off >= 1; off >>= 1) {
#pragma unroll
    for (int i = 0; i < 8; ++i)
      srow[i] += __shfl_xor(srow[i], off, 32);
  }
  if (ln == 0) {
    const float vb = Vb[0];
#pragma unroll
    for (int i = 0; i < 8; ++i)
      scores[m0 + 8 * hi + i] = srow[i] + vb;
  }
}

// ---------------------------------------------------------------------------
// Embedding gather: gi_in[b, 0:512] = emb_table[x[b]]
// ---------------------------------------------------------------------------
__global__ __launch_bounds__(256) void embed_gather(
    const int* __restrict__ x, const float* __restrict__ emb,
    float* __restrict__ gi_in) {
  const int b = blockIdx.x;
  const long row = x[b];
  for (int e = threadIdx.x; e < 512; e += 256)
    gi_in[(long)b * 1024 + e] = emb[row * 512 + e];
}

// ---------------------------------------------------------------------------
// Softmax over S=256 + context GEMV: ctx[b,e] = sum_s alpha[b,s]*enc[b,s,e]
// One block per batch row. ctx_out points at gi_in + 512 (ld = 1024).
// ---------------------------------------------------------------------------
__global__ __launch_bounds__(256) void softmax_context(
    const float* __restrict__ scores, const float* __restrict__ enc,
    float* __restrict__ ctx_out) {
  __shared__ float red[256];
  __shared__ float alpha[256];
  const int b = blockIdx.x;
  const int tid = threadIdx.x;

  const float s = scores[b * 256 + tid];
  red[tid] = s;
  __syncthreads();
  for (int off = 128; off >= 1; off >>= 1) {
    if (tid < off) red[tid] = fmaxf(red[tid], red[tid + off]);
    __syncthreads();
  }
  const float mx = red[0];
  __syncthreads();
  const float ev = __expf(s - mx);
  red[tid] = ev;
  __syncthreads();
  for (int off = 128; off >= 1; off >>= 1) {
    if (tid < off) red[tid] += red[tid + off];
    __syncthreads();
  }
  alpha[tid] = ev / red[0];
  __syncthreads();

  const float* __restrict__ encb = enc + (long)b * 256 * 512;
  for (int e0 = tid; e0 < 512; e0 += 256) {
    float acc = 0.0f;
    for (int s2 = 0; s2 < 256; ++s2)
      acc += alpha[s2] * encb[(long)s2 * 512 + e0];
    ctx_out[(long)b * 1024 + e0] = acc;
  }
}

// ---------------------------------------------------------------------------
// GRU gate combine (PyTorch order r,z,n): h_new = (1-z)*n + z*h
// ---------------------------------------------------------------------------
__global__ __launch_bounds__(256) void gru_combine(
    const float* __restrict__ gi, const float* __restrict__ gh,
    const float* __restrict__ h, float* __restrict__ hnew) {
  const int b = blockIdx.x;
  for (int j = threadIdx.x; j < 512; j += 256) {
    const long base = (long)b * 1536 + j;
    const float r = 1.0f / (1.0f + __expf(-(gi[base] + gh[base])));
    const float z = 1.0f / (1.0f + __expf(-(gi[base + 512] + gh[base + 512])));
    const float n = tanhf(gi[base + 1024] + r * gh[base + 1024]);
    hnew[(long)b * 512 + j] = (1.0f - z) * n + z * h[(long)b * 512 + j];
  }
}

// ---------------------------------------------------------------------------
// Host-side orchestration
// ---------------------------------------------------------------------------
extern "C" void kernel_launch(void* const* d_in, const int* in_sizes, int n_in,
                              void* d_out, int out_size, void* d_ws,
                              size_t ws_size, hipStream_t stream) {
  (void)in_sizes; (void)n_in; (void)out_size;
  const int*   x      = (const int*)d_in[0];
  const float* hidden = (const float*)d_in[1];   // [1,64,512] -> [64,512]
  const float* enc    = (const float*)d_in[2];   // [64,256,512]
  const float* emb    = (const float*)d_in[3];   // [32000,512]
  const float* W_w    = (const float*)d_in[4];   // [512,512]
  const float* W_b    = (const float*)d_in[5];
  const float* U_w    = (const float*)d_in[6];   // [512,512]
  const float* U_b    = (const float*)d_in[7];
  const float* V_w    = (const float*)d_in[8];   // [1,512]
  const float* V_b    = (const float*)d_in[9];
  const float* W_ih   = (const float*)d_in[10];  // [1536,1024]
  const float* W_hh   = (const float*)d_in[11];  // [1536,512]
  const float* b_ih   = (const float*)d_in[12];
  const float* b_hh   = (const float*)d_in[13];
  const float* fc_w   = (const float*)d_in[14];  // [32000,512]
  const float* fc_b   = (const float*)d_in[15];

  float* pred = (float*)d_out;                      // [64,32000]
  float* hnew = pred + (size_t)64 * 32000;          // [1,64,512]

  // Scratch layout (~1.22 MB). Fall back to the prediction region of d_out
  // (dead until the final GEMM) if ws is too small.
  const size_t need = (size_t)(64 * 1024 + 64 * 512 + 64 * 256 +
                               2 * 64 * 1536) * sizeof(float);
  float* scr    = (ws_size >= need) ? (float*)d_ws : pred;
  float* gi_in  = scr;                      // [64,1024] = [embedded | context]
  float* Uh     = gi_in + 64 * 1024;        // [64,512]
  float* scores = Uh + 64 * 512;            // [64,256]
  float* gi     = scores + 64 * 256;        // [64,1536]
  float* gh     = gi + 64 * 1536;           // [64,1536]

  // 1) embedded = emb_table[x]
  embed_gather<<<64, 256, 0, stream>>>(x, emb, gi_in);

  // 2) Uh = h @ U_w^T + U_b            (M=64, N=512, K=512)
  gemm_wmma_f16x4<<<dim3(8, 4), 32, 0, stream>>>(hidden, 512, U_w, 512, U_b,
                                                 Uh, 512, 512);

  // 3) scores = V . tanh(enc@Ww^T + Wb + Uh) + Vb   (the 8.6 GFLOP GEMM, fused)
  attn_scores_wmma<<<1024, 32, 0, stream>>>(enc, W_w, W_b, Uh, V_w, V_b,
                                            scores);

  // 4) alpha = softmax(scores); context = alpha @ enc  -> gi_in[:,512:1024]
  softmax_context<<<64, 256, 0, stream>>>(scores, enc, gi_in + 512);

  // 5) gi = [embedded|context] @ W_ih^T + b_ih   (M=64, N=1536, K=1024)
  gemm_wmma_f16x4<<<dim3(24, 4), 32, 0, stream>>>(gi_in, 1024, W_ih, 1024,
                                                  b_ih, gi, 1536, 1024);
  //    gh = h @ W_hh^T + b_hh                   (M=64, N=1536, K=512)
  gemm_wmma_f16x4<<<dim3(24, 4), 32, 0, stream>>>(hidden, 512, W_hh, 512,
                                                  b_hh, gh, 1536, 512);

  // 6) GRU gate combine -> h_new (second output, also input to fc)
  gru_combine<<<64, 256, 0, stream>>>(gi, gh, hidden, hnew);

  // 7) prediction = h_new @ fc_w^T + fc_b       (M=64, N=32000, K=512)
  gemm_wmma_f16x4<<<dim3(500, 4), 32, 0, stream>>>(hnew, 512, fc_w, 512, fc_b,
                                                   pred, 32000, 512);
}
